// LRSHA_9981503996443
// MI455X (gfx1250) — matrix operand order repaired and checked by reading, not compiled, not run on hardware
//
#include <hip/hip_runtime.h>

// ---- problem constants (from reference) ----
constexpr int kB    = 16;
constexpr int kDIM  = 256;
constexpr int kQK   = 64;
constexpr int kPDIM = 128;
constexpr int kNP   = 4096;   // H*W = 64*64
constexpr int kM    = 1024;   // (H/2)*(W/2)
constexpr int kCY   = 384;    // 2*QK + DIM
constexpr float kEPS   = 1e-5f;
constexpr float kSCALE = 0.125f; // QK^-0.5

// ---- WMMA types / helpers ----
typedef __attribute__((ext_vector_type(16))) __bf16 v16bf;
typedef __attribute__((ext_vector_type(8)))  float  v8f;

union FragA { unsigned u[8]; v16bf v; };
union FragC { float f[8]; v8f v; };

static __device__ __forceinline__ v8f wmma_bf16(v16bf a, v16bf b, v8f c) {
  return __builtin_amdgcn_wmma_f32_16x16x32_bf16(false, a, false, b, (short)0, c, false, false);
}

// float -> bf16 (round to nearest even), returned as raw u16
static __device__ __forceinline__ unsigned short f2bf(float f) {
  unsigned u = __float_as_uint(f);
  unsigned r = u + 0x7FFFu + ((u >> 16) & 1u);
  return (unsigned short)(r >> 16);
}

// A-fragment K offset for element pair i (0..7), contiguous order k = 16*lhi + 2*i (B frags use the same)
static __device__ __forceinline__ int koffA(int i, int lhi) {
  return (lhi ? 16 : 0) + (i << 1);
}

// LDS byte offset of a shared-memory pointer (generic addr low 32 bits)
static __device__ __forceinline__ unsigned lds_off(const void* p) {
  return (unsigned)(size_t)p;
}

// =====================================================================
// Kernel 1: GroupNorm(1 group) statistics per batch
// =====================================================================
__global__ __launch_bounds__(256) void gn_stats_kernel(const float* __restrict__ x,
                                                       float* __restrict__ stats) {
  __shared__ double s1[256];
  __shared__ double s2[256];
  int b = blockIdx.x, t = threadIdx.x;
  const float* xb = x + (size_t)b * kDIM * kNP;
  double a = 0.0, c = 0.0;
  for (int i = t; i < kDIM * kNP; i += 256) {
    float v = xb[i];
    a += (double)v;
    c += (double)v * (double)v;
  }
  s1[t] = a; s2[t] = c; __syncthreads();
  for (int off = 128; off > 0; off >>= 1) {
    if (t < off) { s1[t] += s1[t + off]; s2[t] += s2[t + off]; }
    __syncthreads();
  }
  if (t == 0) {
    double n = (double)(kDIM * kNP);
    double mu = s1[0] / n;
    double var = s2[0] / n - mu * mu;
    stats[2 * b]     = (float)mu;
    stats[2 * b + 1] = rsqrtf((float)var + kEPS);
  }
}

// =====================================================================
// Kernel 2: apply GN, emit bf16 activations
// =====================================================================
__global__ __launch_bounds__(256) void gn_apply_kernel(const float* __restrict__ x,
                                                       const float* __restrict__ g,
                                                       const float* __restrict__ bta,
                                                       const float* __restrict__ stats,
                                                       unsigned short* __restrict__ xn) {
  int idx = blockIdx.x * 256 + threadIdx.x;     // < 16*256*4096 = 2^24
  int b = idx >> 20;
  int c = (idx >> 12) & 255;
  float mu = stats[2 * b], ri = stats[2 * b + 1];
  float v = (x[idx] - mu) * ri * g[c] + bta[c];
  xn[idx] = f2bf(v);
}

// =====================================================================
// Kernel 3: conv_in 3x3 (256 -> 384) as implicit GEMM + BN.
// Block tile: 64 pixels (one image row) x 64 output channels, 4 waves,
// each wave 32x32 via 2x2 WMMA 16x16x32 bf16 tiles. Epilogue also writes
// q channels transposed (qT[b][n][c], bf16) for the attention kernel.
// =====================================================================
__global__ __launch_bounds__(128) void conv_in_kernel(const unsigned short* __restrict__ xn,
                                                      const float* __restrict__ w_in,
                                                      const float* __restrict__ bg,
                                                      const float* __restrict__ bbi,
                                                      const float* __restrict__ bm,
                                                      const float* __restrict__ bv,
                                                      float* __restrict__ y,
                                                      unsigned short* __restrict__ qT) {
  __shared__ unsigned short Xl[64][32];   // [pixel][k]   (A pairs contiguous)
  __shared__ unsigned short Wt[64][32];   // [oc][k]      (B pairs contiguous)

  int bx   = blockIdx.x;
  int oct  = bx % 6;                 // 6 oc tiles of 64 (384)
  int tile = bx / 6;                 // b*64 + h
  int b = tile >> 6, h = tile & 63;
  int oc0 = oct * 64;

  int t = threadIdx.x, lane = t & 31, wid = t >> 5;
  int wpix = wid >> 1, woc = wid & 1;
  int lrow = lane & 15, lhi = lane >> 4;

  FragC acc[2][2];
  for (int pi = 0; pi < 2; ++pi)
    for (int oi = 0; oi < 2; ++oi)
      for (int r = 0; r < 8; ++r) acc[pi][oi].f[r] = 0.0f;

  const unsigned short* xb = xn + (size_t)b * kDIM * kNP;

  for (int pos = 0; pos < 9; ++pos) {
    int kh = pos / 3 - 1, kw = pos % 3 - 1;
    int hh = h + kh;
    for (int icb = 0; icb < 8; ++icb) {
      int ic0 = icb * 32;
      __syncthreads();
      // stage input patch (zero-padded halo)
      for (int i = 0; i < 16; ++i) {
        int idx = t + i * 128;              // 2048 elements
        int p = idx & 63, kk = idx >> 6;
        int ww = p + kw;
        unsigned short val = 0;
        if (hh >= 0 && hh < 64 && ww >= 0 && ww < 64)
          val = xb[(size_t)(ic0 + kk) * kNP + hh * 64 + ww];
        Xl[p][kk] = val;
      }
      // stage weights (oc-major so B-frag pairs along k are contiguous)
      for (int i = 0; i < 16; ++i) {
        int idx = t + i * 128;
        int ocl = idx >> 5, kk = idx & 31;
        Wt[ocl][kk] =
            f2bf(w_in[(size_t)(oc0 + ocl) * (kDIM * 9) + (size_t)(ic0 + kk) * 9 +
                      (kh + 1) * 3 + (kw + 1)]);
      }
      __syncthreads();

      FragA afr[2], bfr[2];
      for (int pi = 0; pi < 2; ++pi) {
        int p = wpix * 32 + pi * 16 + lrow;
        for (int i = 0; i < 8; ++i)
          afr[pi].u[i] = *(const unsigned*)&Xl[p][koffA(i, lhi)];
      }
      for (int oi = 0; oi < 2; ++oi) {
        int ocl = woc * 32 + oi * 16 + lrow;
        for (int j = 0; j < 8; ++j)
          bfr[oi].u[j] = *(const unsigned*)&Wt[ocl][lhi * 16 + 2 * j];
      }
      for (int pi = 0; pi < 2; ++pi)
        for (int oi = 0; oi < 2; ++oi)
          acc[pi][oi].v = wmma_bf16(afr[pi].v, bfr[oi].v, acc[pi][oi].v);
    }
  }

  // epilogue: BN, write y fp32; q channels also transposed bf16
  for (int pi = 0; pi < 2; ++pi)
    for (int oi = 0; oi < 2; ++oi) {
      int oc = oc0 + woc * 32 + oi * 16 + lrow;      // C-tile col = oc
      float s  = bg[oc] * rsqrtf(bv[oc] + kEPS);
      float bi = bbi[oc] - bm[oc] * s;
      for (int r = 0; r < 8; ++r) {
        int p = wpix * 32 + pi * 16 + r + 8 * lhi;   // C-tile row = pixel
        int n = h * 64 + p;
        float val = acc[pi][oi].f[r] * s + bi;
        y[((size_t)b * kCY + oc) * kNP + n] = val;
        if (oc < kQK)
          qT[((size_t)b * kNP + n) * kQK + oc] = f2bf(val);
      }
    }
}

// =====================================================================
// Kernel 4: depthwise 2x2 stride-2 convs for k and v + BN, bf16 output
// kT:   (B, 1024, 64) m-major  (so attention K tiles are contiguous 4KB)
// vbuf: (B, 128, 1024) c-major
// =====================================================================
__global__ __launch_bounds__(256) void dw_kernel(const float* __restrict__ y,
                                                 const float* __restrict__ wk,
                                                 const float* __restrict__ kg, const float* __restrict__ kb,
                                                 const float* __restrict__ km, const float* __restrict__ kv,
                                                 const float* __restrict__ wv,
                                                 const float* __restrict__ vg, const float* __restrict__ vb,
                                                 const float* __restrict__ vm, const float* __restrict__ vvv,
                                                 unsigned short* __restrict__ kT,
                                                 unsigned short* __restrict__ vbuf) {
  int idx = blockIdx.x * 256 + threadIdx.x;   // < 16*192*1024
  int m    = idx & 1023;
  int c192 = (idx >> 10) % 192;
  int b    = idx / (192 * 1024);
  int mh = m >> 5, mw = m & 31;

  int c, sc;
  const float *w, *g, *bt, *mm, *vr;
  unsigned short* out;
  if (c192 < kQK) {
    c = c192; sc = kQK + c;
    w = wk + c * 4; g = kg; bt = kb; mm = km; vr = kv;
    out = kT + ((size_t)b * kM + m) * kQK + c;         // transposed
  } else {
    c = c192 - kQK; sc = 2 * kQK + c;
    w = wv + c * 4; g = vg; bt = vb; mm = vm; vr = vvv;
    out = vbuf + ((size_t)b * kPDIM + c) * kM + m;
  }
  const float* src = y + ((size_t)b * kCY + sc) * kNP;
  float s = 0.0f;
  for (int dh = 0; dh < 2; ++dh)
    for (int dw2 = 0; dw2 < 2; ++dw2)
      s += src[(2 * mh + dh) * 64 + 2 * mw + dw2] * w[dh * 2 + dw2];
  float sg = g[c] * rsqrtf(vr[c] + kEPS);
  *out = f2bf((s - mm[c]) * sg + bt[c]);
}

// =====================================================================
// Async global->LDS staging of one 32-key K/V tile (double-buffered).
// K tile: contiguous 4KB slab of kT (m-major). V tile: 128 rows x 64B.
// Uses GLOBAL_LOAD_ASYNC_TO_LDS_B128 (GVS mode), tracked by ASYNCcnt.
// =====================================================================
static __device__ __forceinline__ void stage_kv_async(unsigned long long kTb,
                                                      unsigned long long vBb,
                                                      unsigned ldsK, unsigned ldsV,
                                                      int t, int m0) {
  // K: 4096 bytes, 128 threads x 2 chunks of 16B (offset imm applies to both
  // the global and LDS addresses of the async copy)
  unsigned gK = (unsigned)(m0 * (kQK * 2) + t * 16);
  unsigned lK = ldsK + (unsigned)(t * 16);
  asm volatile("global_load_async_to_lds_b128 %0, %1, %2\n\t"
               "global_load_async_to_lds_b128 %0, %1, %2 offset:2048"
               :: "v"(lK), "v"(gK), "s"(kTb)
               : "memory");
  // V: 128 rows (stride 2048B in global) x 64B -> contiguous 8KB in LDS
  for (int i = 0; i < 4; ++i) {
    int idx = t + i * 128;                 // 512 chunks of 16B
    unsigned lV = ldsV + (unsigned)(idx * 16);
    unsigned gV = (unsigned)((idx >> 2) * (kM * 2) + (idx & 3) * 16 + m0 * 2);
    asm volatile("global_load_async_to_lds_b128 %0, %1, %2"
                 :: "v"(lV), "v"(gV), "s"(vBb)
                 : "memory");
  }
}

// =====================================================================
// Kernel 5: flash attention. 1 wave = 16 queries, block = 4 waves = 64.
// Double-buffered async K/V staging overlapped with compute. S = Q^T K via
// chained bf16 WMMA (k=64); online softmax with shfl width-16; P re-laid
// out C->A through LDS; O (16 x 128) accumulated with 8 WMMAs per tile.
// =====================================================================
__global__ __launch_bounds__(128) void attn_kernel(const unsigned short* __restrict__ qT,
                                                   const unsigned short* __restrict__ kT,
                                                   const unsigned short* __restrict__ vbuf,
                                                   unsigned short* __restrict__ obuf) {
  __shared__ __align__(16) unsigned short KtB[2][32][64];   // [m][c]
  __shared__ __align__(16) unsigned short VlB[2][128][32];  // [c][m]
  __shared__ unsigned short Pl[4][16][32];                  // per-wave P bounce

  int bx = blockIdx.x;
  int b  = bx >> 6;                  // 64 n-tiles of 64 per batch
  int n0blk = (bx & 63) * 64;
  int t = threadIdx.x, lane = t & 31, wid = t >> 5;
  int lrow = lane & 15, lhi = lane >> 4;
  int n0 = n0blk + wid * 16;

  unsigned long long kTb = (unsigned long long)(kT + (size_t)b * kM * kQK);
  unsigned long long vBb = (unsigned long long)(vbuf + (size_t)b * kPDIM * kM);
  unsigned ldsK0 = lds_off(&KtB[0][0][0]), ldsK1 = lds_off(&KtB[1][0][0]);
  unsigned ldsV0 = lds_off(&VlB[0][0][0]), ldsV1 = lds_off(&VlB[1][0][0]);

  // preload Q A-fragments (16 n x 64 c, two k=32 halves)
  FragA aq0, aq1;
  {
    const unsigned short* qrow = qT + ((size_t)b * kNP + n0 + lrow) * kQK;
    for (int i = 0; i < 8; ++i) {
      int ko = koffA(i, lhi);
      aq0.u[i] = *(const unsigned*)&qrow[ko];
      aq1.u[i] = *(const unsigned*)&qrow[32 + ko];
    }
  }

  FragC Oc[8];
  for (int ct = 0; ct < 8; ++ct)
    for (int r = 0; r < 8; ++r) Oc[ct].f[r] = 0.0f;
  float rowM[8], rowL[8];
  for (int r = 0; r < 8; ++r) { rowM[r] = -1e30f; rowL[r] = 0.0f; }

  // prologue: stage tile 0 into buffer 0
  stage_kv_async(kTb, vBb, ldsK0, ldsV0, t, 0);

  for (int it = 0; it < kM / 32; ++it) {
    int p = it & 1;
    // wait for this tile's async copies (per-wave), then block barrier
    asm volatile("s_wait_asynccnt 0" ::: "memory");
    __syncthreads();
    // kick off next tile into the other buffer (safe: all waves finished
    // reading it one iteration ago, enforced by the barrier above)
    if (it + 1 < kM / 32) {
      stage_kv_async(kTb, vBb, (p ? ldsK0 : ldsK1), (p ? ldsV0 : ldsV1),
                     t, (it + 1) * 32);
    }

    // S = Q^T K  (two 16x16 m-tiles, k=64 as 2 chained k=32 WMMAs)
    FragC s[2];
    for (int mt = 0; mt < 2; ++mt) {
      for (int r = 0; r < 8; ++r) s[mt].f[r] = 0.0f;
      int mcol = mt * 16 + lrow;
      FragA bk0, bk1;
      for (int j = 0; j < 8; ++j) {
        bk0.u[j] = *(const unsigned*)&KtB[p][mcol][lhi * 16 + 2 * j];
        bk1.u[j] = *(const unsigned*)&KtB[p][mcol][32 + lhi * 16 + 2 * j];
      }
      s[mt].v = wmma_bf16(aq0.v, bk0.v, s[mt].v);
      s[mt].v = wmma_bf16(aq1.v, bk1.v, s[mt].v);
    }
    for (int mt = 0; mt < 2; ++mt)
      for (int r = 0; r < 8; ++r) s[mt].f[r] *= kSCALE;

    // online softmax (rows 0..7 live in lanes 0..15; 8..15 in lanes 16..31)
    for (int r = 0; r < 8; ++r) {
      float v = fmaxf(s[0].f[r], s[1].f[r]);
      for (int off = 8; off >= 1; off >>= 1) v = fmaxf(v, __shfl_xor(v, off, 16));
      float nm = fmaxf(rowM[r], v);
      float p0 = __expf(s[0].f[r] - nm);
      float p1 = __expf(s[1].f[r] - nm);
      s[0].f[r] = p0; s[1].f[r] = p1;
      float rs = p0 + p1;
      for (int off = 8; off >= 1; off >>= 1) rs += __shfl_xor(rs, off, 16);
      float alpha = __expf(rowM[r] - nm);
      rowL[r] = rowL[r] * alpha + rs;
      rowM[r] = nm;
      for (int ct = 0; ct < 8; ++ct) Oc[ct].f[r] *= alpha;
    }

    // P: C layout -> A layout via per-wave LDS bounce
    for (int mt = 0; mt < 2; ++mt)
      for (int r = 0; r < 8; ++r)
        Pl[wid][r + 8 * lhi][mt * 16 + lrow] = f2bf(s[mt].f[r]);
    __syncthreads();
    FragA pa;
    for (int i = 0; i < 8; ++i)
      pa.u[i] = *(const unsigned*)&Pl[wid][lrow][koffA(i, lhi)];

    // O += P x V^T   (8 c-tiles of 16)
    for (int ct = 0; ct < 8; ++ct) {
      int ccol = ct * 16 + lrow;
      FragA bvf;
      for (int j = 0; j < 8; ++j)
        bvf.u[j] = *(const unsigned*)&VlB[p][ccol][lhi * 16 + 2 * j];
      Oc[ct].v = wmma_bf16(pa.v, bvf.v, Oc[ct].v);
    }
  }

  // finalize: O /= l, store bf16 c-major (B, 128, 4096)
  for (int r = 0; r < 8; ++r) {
    float inv = rowL[r] > 0.0f ? 1.0f / rowL[r] : 0.0f;
    int n = n0 + r + 8 * lhi;
    for (int ct = 0; ct < 8; ++ct) {
      int c = ct * 16 + lrow;
      obuf[((size_t)b * kPDIM + c) * kNP + n] = f2bf(Oc[ct].f[r] * inv);
    }
  }
}

// =====================================================================
// Kernel 6: 1x1 conv (256 -> 256) GEMM over relu(cat(o,u)) + BN -> d_out
// Block tile 64 oc x 64 px, 4 waves x (2x2) WMMA tiles. A = w_out (direct
// global), B = activations staged px-major in LDS with fused relu.
// =====================================================================
__global__ __launch_bounds__(128) void conv_out_kernel(const unsigned short* __restrict__ obuf,
                                                       const float* __restrict__ y,
                                                       const float* __restrict__ w_out,
                                                       const float* __restrict__ bg,
                                                       const float* __restrict__ bbo,
                                                       const float* __restrict__ bm,
                                                       const float* __restrict__ bv,
                                                       float* __restrict__ out) {
  __shared__ unsigned short Bl[64][32];   // [pixel][ic]

  int bx  = blockIdx.x;
  int oct = bx & 3;
  int pt  = bx >> 2;                      // 1024 pixel tiles (64 per batch)
  int b   = pt >> 6;
  int n0  = (pt & 63) * 64;
  int oc0 = oct * 64;

  int t = threadIdx.x, lane = t & 31, wid = t >> 5;
  int woc = wid >> 1, wpx = wid & 1;
  int lrow = lane & 15, lhi = lane >> 4;

  FragC acc[2][2];
  for (int oi = 0; oi < 2; ++oi)
    for (int pi = 0; pi < 2; ++pi)
      for (int r = 0; r < 8; ++r) acc[oi][pi].f[r] = 0.0f;

  for (int icb = 0; icb < 8; ++icb) {
    int ic0 = icb * 32;
    __syncthreads();
    for (int i = 0; i < 16; ++i) {
      int idx = t + i * 128;
      int p = idx & 63, kk = idx >> 6;
      int ic = ic0 + kk;
      unsigned short uv;
      if (ic < kPDIM) {
        unsigned short o = obuf[((size_t)b * kPDIM + ic) * kNP + n0 + p];
        uv = (o & 0x8000u) ? (unsigned short)0 : o;   // relu on bf16
      } else {
        float f = y[((size_t)b * kCY + (2 * kQK + kPDIM) + (ic - kPDIM)) * kNP + n0 + p];
        uv = f2bf(fmaxf(f, 0.0f));
      }
      Bl[p][kk] = uv;
    }
    __syncthreads();

    FragA aw[2], bp[2];
    for (int oi = 0; oi < 2; ++oi) {
      int oc = oc0 + woc * 32 + oi * 16 + lrow;
      const float* wrow = w_out + (size_t)oc * kDIM + ic0;
      for (int i = 0; i < 8; ++i) {
        int ko = koffA(i, lhi);
        aw[oi].u[i] = (unsigned)f2bf(wrow[ko]) | ((unsigned)f2bf(wrow[ko + 1]) << 16);
      }
    }
    for (int pi = 0; pi < 2; ++pi) {
      int p = wpx * 32 + pi * 16 + lrow;
      for (int j = 0; j < 8; ++j)
        bp[pi].u[j] = *(const unsigned*)&Bl[p][lhi * 16 + 2 * j];
    }
    for (int oi = 0; oi < 2; ++oi)
      for (int pi = 0; pi < 2; ++pi)
        acc[oi][pi].v = wmma_bf16(aw[oi].v, bp[pi].v, acc[oi][pi].v);
  }

  for (int oi = 0; oi < 2; ++oi)
    for (int pi = 0; pi < 2; ++pi)
      for (int r = 0; r < 8; ++r) {
        int oc = oc0 + woc * 32 + oi * 16 + r + 8 * lhi;   // C row = oc
        int p  = wpx * 32 + pi * 16 + lrow;                // C col = pixel
        float s = bg[oc] * rsqrtf(bv[oc] + kEPS);
        float val = (acc[oi][pi].f[r] - bm[oc]) * s + bbo[oc];
        out[((size_t)b * kDIM + oc) * kNP + n0 + p] = val;
      }
}

// =====================================================================
extern "C" void kernel_launch(void* const* d_in, const int* in_sizes, int n_in,
                              void* d_out, int out_size, void* d_ws, size_t ws_size,
                              hipStream_t stream) {
  (void)in_sizes; (void)n_in; (void)out_size; (void)ws_size;
  const float* x     = (const float*)d_in[0];
  const float* gn_g  = (const float*)d_in[1];
  const float* gn_b  = (const float*)d_in[2];
  const float* w_in  = (const float*)d_in[3];
  const float* big   = (const float*)d_in[4];
  const float* bib   = (const float*)d_in[5];
  const float* bim   = (const float*)d_in[6];
  const float* biv   = (const float*)d_in[7];
  const float* w_k   = (const float*)d_in[8];
  const float* bkg   = (const float*)d_in[9];
  const float* bkb   = (const float*)d_in[10];
  const float* bkm   = (const float*)d_in[11];
  const float* bkv   = (const float*)d_in[12];
  const float* w_v   = (const float*)d_in[13];
  const float* bvg   = (const float*)d_in[14];
  const float* bvb   = (const float*)d_in[15];
  const float* bvm   = (const float*)d_in[16];
  const float* bvv   = (const float*)d_in[17];
  const float* w_out = (const float*)d_in[18];
  const float* bog   = (const float*)d_in[19];
  const float* bob   = (const float*)d_in[20];
  const float* bom   = (const float*)d_in[21];
  const float* bov   = (const float*)d_in[22];

  char* ws = (char*)d_ws;
  size_t off = 0;
  float* stats = (float*)(ws + off);                  off += 256;
  unsigned short* xn   = (unsigned short*)(ws + off); off += (size_t)kB * kDIM * kNP * 2;
  float*          yb   = (float*)(ws + off);          off += (size_t)kB * kCY * kNP * 4;
  unsigned short* qT   = (unsigned short*)(ws + off); off += (size_t)kB * kNP * kQK * 2;
  unsigned short* kT   = (unsigned short*)(ws + off); off += (size_t)kB * kM * kQK * 2;
  unsigned short* vbuf = (unsigned short*)(ws + off); off += (size_t)kB * kPDIM * kM * 2;
  unsigned short* obuf = (unsigned short*)(ws + off); off += (size_t)kB * kPDIM * kNP * 2;

  gn_stats_kernel<<<kB, 256, 0, stream>>>(x, stats);
  gn_apply_kernel<<<(kB * kDIM * kNP) / 256, 256, 0, stream>>>(x, gn_g, gn_b, stats, xn);
  conv_in_kernel<<<6 * kB * 64, 128, 0, stream>>>(xn, w_in, big, bib, bim, biv, yb, qT);
  dw_kernel<<<(kB * (kQK + kPDIM) * kM) / 256, 256, 0, stream>>>(
      yb, w_k, bkg, bkb, bkm, bkv, w_v, bvg, bvb, bvm, bvv, kT, vbuf);
  attn_kernel<<<kB * (kNP / 64), 128, 0, stream>>>(qT, kT, vbuf, obuf);
  conv_out_kernel<<<(kB * kNP / 64) * (kDIM / 64), 128, 0, stream>>>(
      obuf, yb, w_out, bog, bob, bom, bov, (float*)d_out);
}
